// NoseNet_55430847922252
// MI455X (gfx1250) — matrix operand order repaired and checked by reading, not compiled, not run on hardware
//
#include <hip/hip_runtime.h>
#include <hip/hip_bf16.h>
#include <stdint.h>

#define NB_FEATURES 512
#define MB_DIM      10240
#define NB_CLASSES  100
#define BATCH       8192
#define HASH_LEN    32

typedef __attribute__((ext_vector_type(16))) __bf16 v16bf;
typedef __attribute__((ext_vector_type(8)))  __bf16 v8bf;
typedef __attribute__((ext_vector_type(8)))  float  v8f;

__device__ __forceinline__ unsigned short f32_to_bf16_bits(float f) {
  unsigned int u = __float_as_uint(f);
  unsigned int r = u + 0x7FFFu + ((u >> 16) & 1u);   // round-to-nearest-even
  return (unsigned short)(r >> 16);
}

// ---------------------------------------------------------------------------
// Kernel 1: convert x and w1 to bf16 (w1 is binary {0,1} -> exact in bf16)
// ---------------------------------------------------------------------------
__global__ void nose_cvt_kernel(const float* __restrict__ x,
                                const float* __restrict__ w1,
                                unsigned short* __restrict__ xb,
                                unsigned short* __restrict__ w1b) {
  size_t i = (size_t)blockIdx.x * blockDim.x + threadIdx.x;
  const size_t nx = (size_t)BATCH * NB_FEATURES;
  const size_t nw = (size_t)MB_DIM * NB_FEATURES;
  if (i < nx) {
    xb[i] = f32_to_bf16_bits(x[i]);
  } else if (i < nx + nw) {
    size_t j = i - nx;
    w1b[j] = f32_to_bf16_bits(w1[j]);
  }
}

// ---------------------------------------------------------------------------
// Kernel 2: H[b,m] = sum_k Xb[b,k] * W1b[m,k]   (N-T GEMM, bf16 WMMA)
// Block tile 128x128, K-step 64, 256 threads = 8 waves in a 2x4 grid,
// each wave owns a 64x32 sub-tile = 4x2 v_wmma_f32_16x16x32_bf16 accumulators.
// ---------------------------------------------------------------------------
#define BM 128
#define BN 128
#define BK 64
#define LDT 72   // padded LDS row stride (bf16 elements) to spread banks

__global__ __launch_bounds__(256)
void nose_gemm_kernel(const __bf16* __restrict__ Xb,
                      const __bf16* __restrict__ W1b,
                      float* __restrict__ H) {
  __shared__ __align__(16) __bf16 sA[BM * LDT];
  __shared__ __align__(16) __bf16 sB[BN * LDT];

  const int tid  = threadIdx.x;
  const int lane = tid & 31;
  const int wave = tid >> 5;
  const int wr   = wave >> 2;      // 0..1 : 64-row group
  const int wc   = wave & 3;       // 0..3 : 32-col group
  const int lrow = lane & 15;
  const int lhi  = lane >> 4;      // 0/1 : high half-wave

  const int b0 = blockIdx.y * BM;
  const int n0 = blockIdx.x * BN;

  v8f acc[4][2];
#pragma unroll
  for (int i = 0; i < 4; ++i)
#pragma unroll
    for (int j = 0; j < 2; ++j)
      acc[i][j] = (v8f){0.f, 0.f, 0.f, 0.f, 0.f, 0.f, 0.f, 0.f};

  for (int k0 = 0; k0 < NB_FEATURES; k0 += BK) {
    // Stage 128x64 bf16 tiles of X and W1 into LDS (16B chunks, coalesced).
#pragma unroll
    for (int t = 0; t < 4; ++t) {
      int idx = t * 256 + tid;           // 0..1023
      int row = idx >> 3;                // 0..127
      int ck  = (idx & 7) * 8;           // element col 0,8,..,56
      *(uint4*)&sA[row * LDT + ck] =
          *(const uint4*)(Xb  + (size_t)(b0 + row) * NB_FEATURES + k0 + ck);
      *(uint4*)&sB[row * LDT + ck] =
          *(const uint4*)(W1b + (size_t)(n0 + row) * NB_FEATURES + k0 + ck);
    }
    __syncthreads();

#pragma unroll
    for (int kk = 0; kk < BK; kk += 32) {
      // B fragment (32x16, K x N): lane holds N = lane&15, K pairs; low
      // half-wave K=0..15, high half-wave K=16..31 -> one contiguous 32B run.
      v16bf bfrag[2];
#pragma unroll
      for (int j = 0; j < 2; ++j) {
        const __bf16* p = &sB[(wc * 32 + j * 16 + lrow) * LDT + kk + lhi * 16];
        v8bf lo = *(const v8bf*)p;
        v8bf hi = *(const v8bf*)(p + 8);
        bfrag[j] = __builtin_shufflevector(lo, hi, 0, 1, 2, 3, 4, 5, 6, 7,
                                           8, 9, 10, 11, 12, 13, 14, 15);
      }
      // A fragment (16x32, M x K): lane holds row M = lane&15; low half-wave
      // K=0..15, high half-wave K=16..31 -> one contiguous 32B run.
#pragma unroll
      for (int i = 0; i < 4; ++i) {
        const __bf16* p = &sA[(wr * 64 + i * 16 + lrow) * LDT + kk + lhi * 16];
        v8bf lo = *(const v8bf*)p;
        v8bf hi = *(const v8bf*)(p + 8);
        v16bf afrag = __builtin_shufflevector(lo, hi, 0, 1, 2, 3, 4, 5, 6, 7,
                                              8, 9, 10, 11, 12, 13, 14, 15);
#pragma unroll
        for (int j = 0; j < 2; ++j) {
          acc[i][j] = __builtin_amdgcn_wmma_f32_16x16x32_bf16(
              false, afrag, false, bfrag[j], (short)0, acc[i][j], false, false);
        }
      }
    }
    __syncthreads();
  }

  // C/D layout: lane holds N = lane&15; VGPR e holds M = e + 8*(lane>=16).
#pragma unroll
  for (int i = 0; i < 4; ++i) {
    int mbase = b0 + wr * 64 + i * 16 + lhi * 8;
#pragma unroll
    for (int j = 0; j < 2; ++j) {
      int n = n0 + wc * 32 + j * 16 + lrow;
      float* o = H + (size_t)mbase * MB_DIM + n;
#pragma unroll
      for (int e = 0; e < 8; ++e)
        o[(size_t)e * MB_DIM] = acc[i][j][e];
    }
  }
}

// ---------------------------------------------------------------------------
// Kernel 3: per-row exact top-32 threshold (bitwise radix-select on monotonic
// sortable keys), compact survivors, tiny sparse matvec against max(w2,0).
// One 256-thread workgroup per batch row.
// ---------------------------------------------------------------------------
#define SEL_MAX 128

__global__ __launch_bounds__(256)
void nose_topk_cls_kernel(const float* __restrict__ H,
                          const float* __restrict__ w2,
                          const float* __restrict__ b2,
                          float* __restrict__ out) {
  __shared__ unsigned int keys[MB_DIM];   // 40 KB
  __shared__ unsigned int red[256];
  __shared__ unsigned int nsel;
  __shared__ unsigned int selIdx[SEL_MAX];
  __shared__ float        selVal[SEL_MAX];

  const int tid = threadIdx.x;
  const int b   = blockIdx.x;
  const float* hrow = H + (size_t)b * MB_DIM;

  // Monotonic float -> uint key (order-preserving bijection).
  for (int i = tid; i < MB_DIM; i += 256) {
    unsigned int u = __float_as_uint(hrow[i]);
    keys[i] = (u & 0x80000000u) ? ~u : (u | 0x80000000u);
  }
  if (tid == 0) nsel = 0;
  __syncthreads();

  // Largest t with count(key >= t) >= 32  ==  32nd-largest key (exact,
  // ties behave identically to reference's h >= th).
  unsigned int thresh = 0;
  for (int bit = 31; bit >= 0; --bit) {
    unsigned int cand = thresh | (1u << bit);
    unsigned int c = 0;
    for (int i = tid; i < MB_DIM; i += 256)
      c += (keys[i] >= cand) ? 1u : 0u;
    red[tid] = c;
    __syncthreads();
    for (int s = 128; s > 0; s >>= 1) {
      if (tid < s) red[tid] += red[tid + s];
      __syncthreads();
    }
    unsigned int total = red[0];
    __syncthreads();
    if (total >= HASH_LEN) thresh = cand;
  }

  // Compact the winners (~32, ties included).
  for (int i = tid; i < MB_DIM; i += 256) {
    unsigned int k = keys[i];
    if (k >= thresh) {
      unsigned int p = atomicAdd(&nsel, 1u);
      if (p < SEL_MAX) {
        selIdx[p] = (unsigned int)i;
        unsigned int ub = (k & 0x80000000u) ? (k ^ 0x80000000u) : ~k;
        selVal[p] = __uint_as_float(ub);
      }
    }
  }
  __syncthreads();
  unsigned int ns = nsel;
  if (ns > SEL_MAX) ns = SEL_MAX;

  // out[b,c] = b2[c] + sum_j val_j * max(w2[c, idx_j], 0)
  for (int c = tid; c < NB_CLASSES; c += 256) {
    float acc = b2[c];
    const float* w2r = w2 + (size_t)c * MB_DIM;
    for (unsigned int j = 0; j < ns; ++j)
      acc = fmaf(selVal[j], fmaxf(w2r[selIdx[j]], 0.0f), acc);
    out[(size_t)b * NB_CLASSES + c] = acc;
  }
}

// ---------------------------------------------------------------------------
extern "C" void kernel_launch(void* const* d_in, const int* in_sizes, int n_in,
                              void* d_out, int out_size, void* d_ws, size_t ws_size,
                              hipStream_t stream) {
  const float* x  = (const float*)d_in[0];   // [8192, 512]
  const float* w1 = (const float*)d_in[1];   // [10240, 512]
  const float* w2 = (const float*)d_in[2];   // [100, 10240]
  const float* b2 = (const float*)d_in[3];   // [100]
  float* out = (float*)d_out;                // [8192, 100]

  // Workspace layout: [ Xb bf16 | W1b bf16 | H f32 ]
  char* ws = (char*)d_ws;
  const size_t xb_bytes  = (size_t)BATCH  * NB_FEATURES * sizeof(unsigned short);
  const size_t w1b_bytes = (size_t)MB_DIM * NB_FEATURES * sizeof(unsigned short);
  unsigned short* xb  = (unsigned short*)ws;
  unsigned short* w1b = (unsigned short*)(ws + xb_bytes);
  float*          H   = (float*)(ws + xb_bytes + w1b_bytes);

  {
    size_t n = (size_t)BATCH * NB_FEATURES + (size_t)MB_DIM * NB_FEATURES;
    dim3 grid((unsigned)((n + 255) / 256));
    nose_cvt_kernel<<<grid, 256, 0, stream>>>(x, w1, xb, w1b);
  }
  {
    dim3 grid(MB_DIM / BN, BATCH / BM);   // 80 x 64 workgroups
    nose_gemm_kernel<<<grid, 256, 0, stream>>>((const __bf16*)xb,
                                               (const __bf16*)w1b, H);
  }
  {
    nose_topk_cls_kernel<<<dim3(BATCH), 256, 0, stream>>>(H, w2, b2, out);
  }
}
